// multiHeadAttention_60954175864970
// MI455X (gfx1250) — matrix operand order repaired and checked by reading, not compiled, not run on hardware
//
#include <hip/hip_runtime.h>
#include <stddef.h>
#include <stdint.h>

// ---------------------------------------------------------------------------
// CDNA5 (gfx1250) bf16 WMMA transformer block:
//   out = softmax_causal((yWq)(yWk)^T / sqrt(d)) (yWv)  @ Wff + bff
// Matmuls via V_WMMA_F32_16X16X32_BF16; tile staging via
// GLOBAL_LOAD_ASYNC_TO_LDS_B128 (ASYNCcnt) with double buffering.
// ---------------------------------------------------------------------------

typedef __attribute__((ext_vector_type(16))) __bf16 v16bf;
typedef __attribute__((ext_vector_type(8)))  __bf16 v8bf;
typedef __attribute__((ext_vector_type(8)))  float  v8f;

#define WMMA_BF16(A, B, C) \
  __builtin_amdgcn_wmma_f32_16x16x32_bf16(false, (A), false, (B), (short)0, (C), false, false)

// Async 16-byte global->LDS copy (per-lane), tracked with ASYNCcnt.
// LDS operand is the low 32 bits of the generic pointer (ISA: LDS aperture
// addresses map addr[31:0] -> LDS offset).
__device__ __forceinline__ void async_cp16(void* lds, const void* g) {
  asm volatile("global_load_async_to_lds_b128 %0, %1, off"
               :: "v"((uint32_t)(uintptr_t)lds), "v"(g)
               : "memory");
}
__device__ __forceinline__ void async_wait0() {
  asm volatile("s_wait_asynccnt 0x0" ::: "memory");
}

// A-matrix fragment (16x32 bf16, MxK). Lane holds row M = lane%16.
// lane<16 : K = [0..8) u [16..24) ; lane>=16 : K = [8..16) u [24..32)
__device__ __forceinline__ v16bf make_afrag(const __bf16* rowPtr, int lane) {
  int kb = (lane < 16) ? 0 : 8;
  v8bf lo = *(const v8bf*)(rowPtr + kb);
  v8bf hi = *(const v8bf*)(rowPtr + kb + 16);
  return __builtin_shufflevector(lo, hi, 0,1,2,3,4,5,6,7,8,9,10,11,12,13,14,15);
}

// B-matrix fragment (32x16 bf16, KxN). Lane holds col N = lane%16.
// colPtr points at that column's K-contiguous storage; lane<16: K = [0..8) u [16..24), lane>=16: K = [8..16) u [24..32).
__device__ __forceinline__ v16bf make_bfrag(const __bf16* colPtr, int lane) {
  int ks = (lane < 16) ? 0 : 8;
  v8bf lo = *(const v8bf*)(colPtr + ks);
  v8bf hi = *(const v8bf*)(colPtr + ks + 16);
  return __builtin_shufflevector(lo, hi, 0,1,2,3,4,5,6,7,8,9,10,11,12,13,14,15);
}

// ---------------------------------------------------------------------------
// Conversions
// ---------------------------------------------------------------------------
__global__ void cvt_f32_bf16(const float* __restrict__ in, __bf16* __restrict__ out, int n) {
  int i = blockIdx.x * blockDim.x + threadIdx.x;
  if (i < n) out[i] = (__bf16)in[i];
}

// in[K][N] (f32) -> out[N][K] (bf16): weight transpose so GEMM B-frags are K-contiguous
__global__ void cvt_transpose_bf16(const float* __restrict__ in, __bf16* __restrict__ out,
                                   int K, int N) {
  int i = blockIdx.x * blockDim.x + threadIdx.x;
  if (i < K * N) {
    int nn = i / K;
    int kk = i - nn * K;
    out[i] = (__bf16)in[(size_t)kk * N + nn];
  }
}

// ---------------------------------------------------------------------------
// GEMM: D[M,N] = A[M,K](bf16,row) x Bt[N,K](bf16, B transposed)  (+ bias, f32 out)
// Block: 256 thr = 8 waves in 2x4 grid; block tile 64x256, wave tile 32x64.
// Double-buffered LDS; tiles staged with async global->LDS copies.
// ---------------------------------------------------------------------------
template <bool OUT_BF16>
__global__ __launch_bounds__(256) void gemm_wmma_bf16(
    const __bf16* __restrict__ A, const __bf16* __restrict__ Bt,
    void* __restrict__ outp, const float* __restrict__ bias,
    int M, int N, int K) {
  __shared__ __align__(16) __bf16 Alds[2][64 * 32];
  __shared__ __align__(16) __bf16 Blds[2][256 * 32];

  const int tid  = threadIdx.x;
  const int lane = tid & 31;
  const int wave = tid >> 5;
  const int wr   = wave >> 2;   // 0..1  (M dir)
  const int wc   = wave & 3;    // 0..3  (N dir)
  const int m0   = blockIdx.y * 64;
  const int n0   = blockIdx.x * 256;
  const int half = lane >> 4;
  const int nco  = lane & 15;

  // staging assignment (per thread): one 16B chunk of A, four 16B chunks of Bt
  const int arow = tid >> 2;
  const int akc  = (tid & 3) * 8;

  auto stage = [&](int k0, int buf) {
    async_cp16(Alds[buf] + arow * 32 + akc,
               A + (size_t)(m0 + arow) * K + k0 + akc);
    const __bf16* src = Bt + (size_t)(n0 + tid) * K + k0;
    __bf16* dst = Blds[buf] + tid * 32;
    async_cp16(dst + 0,  src + 0);
    async_cp16(dst + 8,  src + 8);
    async_cp16(dst + 16, src + 16);
    async_cp16(dst + 24, src + 24);
  };

  v8f acc[2][4] = {};
  stage(0, 0);
  int cur = 0;

  for (int k0 = 0; k0 < K; k0 += 32) {
    async_wait0();        // this wave's async copies into buf[cur] done
    __syncthreads();      // everyone's copies visible; prior readers done
    if (k0 + 32 < K) stage(k0 + 32, cur ^ 1);   // prefetch next tile async

    v16bf af[2], bf[4];
#pragma unroll
    for (int mt = 0; mt < 2; ++mt) {
      int row = wr * 32 + mt * 16 + nco;
      af[mt] = make_afrag(Alds[cur] + row * 32, lane);
    }
#pragma unroll
    for (int nt = 0; nt < 4; ++nt) {
      int col = wc * 64 + nt * 16 + nco;
      bf[nt] = make_bfrag(Blds[cur] + col * 32, lane);
    }
#pragma unroll
    for (int mt = 0; mt < 2; ++mt)
#pragma unroll
      for (int nt = 0; nt < 4; ++nt)
        acc[mt][nt] = WMMA_BF16(af[mt], bf[nt], acc[mt][nt]);
    cur ^= 1;
  }

  // epilogue: C/D layout row = v + 8*(lane/16), col = lane%16
#pragma unroll
  for (int mt = 0; mt < 2; ++mt)
#pragma unroll
    for (int nt = 0; nt < 4; ++nt)
#pragma unroll
      for (int v = 0; v < 8; ++v) {
        int row = m0 + wr * 32 + mt * 16 + v + 8 * half;
        int col = n0 + wc * 64 + nt * 16 + nco;
        float r = acc[mt][nt][v];
        if constexpr (OUT_BF16) {
          ((__bf16*)outp)[(size_t)row * N + col] = (__bf16)r;
        } else {
          ((float*)outp)[(size_t)row * N + col] = r + bias[col];
        }
      }
}

// ---------------------------------------------------------------------------
// Flash attention (causal): grid (n/64, H, B); 128 thr = 4 waves.
// Wave w owns 16 query rows; per key-block of 64: S = Q K^T (WMMA),
// online softmax (width-16 shfl row reductions), O += P V (WMMA).
// K tile staged with async global->LDS; V transposed via scalar LDS stores.
// ---------------------------------------------------------------------------
__global__ __launch_bounds__(128) void attn_flash_bf16(
    const __bf16* __restrict__ qkv,  // [B, n, 3, H, 64] bf16
    __bf16* __restrict__ ctx) {      // [B, n, H*64]     bf16
  constexpr int SEQ = 2048, H = 16;
  const int qblk = blockIdx.x;     // 64-row query block
  const int h    = blockIdx.y;
  const int b    = blockIdx.z;
  const int tid  = threadIdx.x;
  const int lane = tid & 31;
  const int w    = tid >> 5;       // wave 0..3
  const int q0   = qblk * 64;
  const int half = lane >> 4;
  const int nco  = lane & 15;

  __shared__ __align__(16) __bf16 Klds[64 * 64];        // [key][d]
  __shared__ __align__(16) __bf16 Vlds[64 * 64];        // [d][key] (transposed)
  __shared__ __align__(16) __bf16 Plds[4][16 * 64];     // per-wave P tile

  // Q fragments (per-wave, loaded once)
  const int qrow = q0 + w * 16 + nco;
  const size_t qbase = ((size_t)(b * SEQ + qrow) * 3 + 0) * 1024 + h * 64;
  v16bf qf[2];
  qf[0] = make_afrag(qkv + qbase + 0,  lane);
  qf[1] = make_afrag(qkv + qbase + 32, lane);

  v8f O[4] = {};
  float mrow[8], lrow[8];
#pragma unroll
  for (int v = 0; v < 8; ++v) { mrow[v] = -1e30f; lrow[v] = 0.0f; }

  for (int j = 0; j <= qblk; ++j) {
    const int k0 = j * 64;
    // ---- stage K (async, row-major) and V (transposed) into LDS ----
    {
      int key = tid >> 1;
      int dc  = (tid & 1) * 32;
      size_t kb = ((size_t)(b * SEQ + k0 + key) * 3 + 1) * 1024 + h * 64 + dc;
      __bf16* kd = Klds + key * 64 + dc;
      async_cp16(kd + 0,  qkv + kb + 0);
      async_cp16(kd + 8,  qkv + kb + 8);
      async_cp16(kd + 16, qkv + kb + 16);
      async_cp16(kd + 24, qkv + kb + 24);
      size_t vb = ((size_t)(b * SEQ + k0 + key) * 3 + 2) * 1024 + h * 64 + dc;
#pragma unroll
      for (int e = 0; e < 32; e += 8) {
        v8bf vv = *(const v8bf*)(qkv + vb + e);
#pragma unroll
        for (int x = 0; x < 8; ++x) Vlds[(dc + e + x) * 64 + key] = vv[x];
      }
    }
    async_wait0();
    __syncthreads();

    // ---- S = Q K^T : 4 key tiles x 2 K-chunks ----
    v8f S[4] = {};
#pragma unroll
    for (int c = 0; c < 2; ++c)
#pragma unroll
      for (int t = 0; t < 4; ++t) {
        int key = t * 16 + nco;
        v16bf kb = make_bfrag(Klds + key * 64 + c * 32, lane);
        S[t] = WMMA_BF16(qf[c], kb, S[t]);
      }

    // ---- scale, causal mask, online softmax ----
    const bool diag = (j == qblk);
#pragma unroll
    for (int v = 0; v < 8; ++v) {
      const int qr = q0 + w * 16 + v + 8 * half;
      float mx = -1e30f;
#pragma unroll
      for (int t = 0; t < 4; ++t) {
        float s = S[t][v] * 0.125f;   // 1/sqrt(64)
        if (diag && (k0 + t * 16 + nco) > qr) s = -1e30f;
        S[t][v] = s;
        mx = fmaxf(mx, s);
      }
#pragma unroll
      for (int d = 1; d < 16; d <<= 1) mx = fmaxf(mx, __shfl_xor(mx, d, 32));
      float mnew = fmaxf(mrow[v], mx);
      float sc   = __expf(mrow[v] - mnew);
      float rs = 0.0f;
#pragma unroll
      for (int t = 0; t < 4; ++t) {
        float p = __expf(S[t][v] - mnew);
        S[t][v] = p;
        rs += p;
      }
#pragma unroll
      for (int d = 1; d < 16; d <<= 1) rs += __shfl_xor(rs, d, 32);
      lrow[v] = lrow[v] * sc + rs;
      mrow[v] = mnew;
#pragma unroll
      for (int t = 0; t < 4; ++t) O[t][v] *= sc;
    }

    // ---- restage P (D-layout -> row-major LDS -> A-layout frags) ----
#pragma unroll
    for (int t = 0; t < 4; ++t)
#pragma unroll
      for (int v = 0; v < 8; ++v)
        Plds[w][(v + 8 * half) * 64 + t * 16 + nco] = (__bf16)S[t][v];
    // per-wave LDS region: same-wave DS ops are in order, no barrier needed

    // ---- O += P V ----
#pragma unroll
    for (int c = 0; c < 2; ++c) {
      v16bf pa = make_afrag(&Plds[w][nco * 64] + c * 32, lane);
#pragma unroll
      for (int t = 0; t < 4; ++t) {
        v16bf vb = make_bfrag(Vlds + (t * 16 + nco) * 64 + c * 32, lane);
        O[t] = WMMA_BF16(pa, vb, O[t]);
      }
    }
    __syncthreads();
  }

  // ---- epilogue: ctx[b, row, h*64 + d] = O / l ----
#pragma unroll
  for (int t = 0; t < 4; ++t)
#pragma unroll
    for (int v = 0; v < 8; ++v) {
      int row = q0 + w * 16 + v + 8 * half;
      int col = h * 64 + t * 16 + nco;
      ctx[(size_t)(b * SEQ + row) * 1024 + col] = (__bf16)(O[t][v] / lrow[v]);
    }
}

// ---------------------------------------------------------------------------
// Host launcher
// ---------------------------------------------------------------------------
extern "C" void kernel_launch(void* const* d_in, const int* in_sizes, int n_in,
                              void* d_out, int out_size, void* d_ws, size_t ws_size,
                              hipStream_t stream) {
  (void)in_sizes; (void)n_in; (void)out_size; (void)ws_size;
  const float* y    = (const float*)d_in[0];   // [2,2048,1024]
  const float* Wqkv = (const float*)d_in[1];   // [1024,3072]
  const float* Wff  = (const float*)d_in[2];   // [1024,1024]
  const float* bff  = (const float*)d_in[3];   // [1024]
  float* out = (float*)d_out;                  // [2,2048,1024]

  char* ws = (char*)d_ws;
  __bf16* y_bf   = (__bf16*)(ws + 0);          //  8.0 MiB  [4096,1024]
  __bf16* wqkv_t = (__bf16*)(ws + 8388608);    //  6.0 MiB  [3072,1024] (transposed)
  __bf16* wff_t  = (__bf16*)(ws + 14680064);   //  2.0 MiB  [1024,1024] (transposed)
  __bf16* qkv_bf = (__bf16*)(ws + 16777216);   // 24.0 MiB  [4096,3072]
  __bf16* ctx_bf = (__bf16*)(ws + 41943040);   //  8.0 MiB  [4096,1024]

  cvt_f32_bf16<<<(4194304 + 255) / 256, 256, 0, stream>>>(y, y_bf, 4194304);
  cvt_transpose_bf16<<<(3145728 + 255) / 256, 256, 0, stream>>>(Wqkv, wqkv_t, 1024, 3072);
  cvt_transpose_bf16<<<(1048576 + 255) / 256, 256, 0, stream>>>(Wff, wff_t, 1024, 1024);

  // QKV projection: [4096,1024] x [1024,3072] -> bf16 [4096,3072]
  gemm_wmma_bf16<true><<<dim3(12, 64), 256, 0, stream>>>(
      y_bf, wqkv_t, (void*)qkv_bf, nullptr, 4096, 3072, 1024);

  // causal flash attention -> ctx bf16 [4096,1024]
  attn_flash_bf16<<<dim3(32, 16, 2), 128, 0, stream>>>(qkv_bf, ctx_bf);

  // output projection + bias: [4096,1024] x [1024,1024] -> f32 out
  gemm_wmma_bf16<false><<<dim3(4, 64), 256, 0, stream>>>(
      ctx_bf, wff_t, (void*)out, bff, 4096, 1024, 1024);
}
